// UncertainReFineModel_24644522344929
// MI455X (gfx1250) — matrix operand order repaired and checked by reading, not compiled, not run on hardware
//
#include <hip/hip_runtime.h>
#include <hip/hip_bf16.h>

// ---------------------------------------------------------------------------
// CDNA5 (gfx1250) implementation of UncertainReFineModel.
//  - Activations kept as bf16 NHWC rows, channel dims zero-padded to mult of 32.
//  - All MLP layers and 1x1 (pointwise) convs -> WMMA bf16 GEMM with fused
//    bias + ReLU6 + residual-add / mask-select epilogues (compile-time MODE).
//  - Two independent accumulator chains per K-loop (N-tile pair) so the
//    WMMA->WMMA RAW hazard slots are filled with useful work, not v_nops.
//  - Epilogue transposes the WMMA C tile through wave-private LDS so outputs,
//    residuals and mask become contiguous per-lane b128/b8 accesses.
//  - Depthwise 3x3 + BN-affine -> vectorized bf16x8 stencil kernel (Cs templated).
//  - Weights (f32) repacked per launch into zero-padded bf16 [Npad,Kpad],
//    staged in LDS; each wave runs 4 M-tiles to amortize the weight stage.
// ---------------------------------------------------------------------------

typedef __bf16 bf16;
typedef __attribute__((ext_vector_type(16))) __bf16 v16bf;
typedef __attribute__((ext_vector_type(8)))  __bf16 v8bf;
typedef __attribute__((ext_vector_type(8)))  float  v8f;

#define B_   2
#define H_   512
#define W_   512
#define HW_  (H_ * W_)
#define BHW_ (B_ * HW_)   // 524,288 pixels (fits u32 everywhere)

// ---------------------------- mask pipeline --------------------------------

__global__ void umax_kernel(const float* __restrict__ coarse, float* __restrict__ um) {
    unsigned i = blockIdx.x * blockDim.x + threadIdx.x;
    if (i >= (unsigned)BHW_) return;
    unsigned b = i / HW_, hw = i % HW_;
    const float* p = coarse + b * 6u * HW_ + hw;
    float m = p[0];
#pragma unroll
    for (int c = 1; c < 6; ++c) m = fmaxf(m, p[c * HW_]);
    um[i] = 1.0f - m;
}

__global__ void dilate_kernel(const float* __restrict__ um, unsigned char* __restrict__ mask) {
    unsigned i = blockIdx.x * blockDim.x + threadIdx.x;
    if (i >= (unsigned)BHW_) return;
    unsigned b = i / HW_, hw = i % HW_;
    int h = (int)(hw / W_), w = (int)(hw % W_);
    float m = -1e30f;
#pragma unroll
    for (int dy = -1; dy <= 1; ++dy) {
        int hh = h + dy;
        if ((unsigned)hh >= (unsigned)H_) continue;
#pragma unroll
        for (int dx = -1; dx <= 1; ++dx) {
            int ww = w + dx;
            if ((unsigned)ww >= (unsigned)W_) continue;
            m = fmaxf(m, um[b * HW_ + (unsigned)hh * W_ + (unsigned)ww]);
        }
    }
    mask[i] = (m > 0.4f) ? 1 : 0;
}

// -------------------------- pack / repack kernels ---------------------------

__global__ void pack_fm_kernel(const float* __restrict__ feat,
                               const float* __restrict__ coarse,
                               bf16* __restrict__ dst) {
    unsigned i = blockIdx.x * blockDim.x + threadIdx.x;
    unsigned total = (unsigned)BHW_ * 96u;
    if (i >= total) return;
    unsigned pix = i / 96u; int c = (int)(i % 96u);
    unsigned b = pix / HW_, hw = pix % HW_;
    float v = 0.0f;
    if (c < 64)      v = feat  [(b * 64u + (unsigned)c)        * HW_ + hw];
    else if (c < 70) v = coarse[(b * 6u  + (unsigned)(c - 64)) * HW_ + hw];
    dst[i] = (bf16)v;
}

__global__ void pack_mat_kernel(const float* __restrict__ src, int N, int K,
                                bf16* __restrict__ dst, int Np, int Kp) {
    unsigned i = blockIdx.x * blockDim.x + threadIdx.x;
    unsigned total = (unsigned)Np * (unsigned)Kp;
    if (i >= total) return;
    int n = (int)(i / (unsigned)Kp), k = (int)(i % (unsigned)Kp);
    float v = (n < N && k < K) ? src[(unsigned)n * (unsigned)K + (unsigned)k] : 0.0f;
    dst[i] = (bf16)v;
}

__global__ void pack_vec_kernel(const float* __restrict__ src, int N,
                                float* __restrict__ dst, int Np) {
    int i = blockIdx.x * blockDim.x + threadIdx.x;
    if (i >= Np) return;
    dst[i] = (src != nullptr && i < N) ? src[i] : 0.0f;
}

__global__ void pack_dw_kernel(const float* __restrict__ dw, int C,
                               bf16* __restrict__ dst, int Cp) {
    int i = blockIdx.x * blockDim.x + threadIdx.x;
    int total = Cp * 9;
    if (i >= total) return;
    int c = i / 9;
    dst[i] = (c < C) ? (bf16)dw[i] : (bf16)0.0f;
}

// --------------------------- WMMA GEMM kernel -------------------------------
// Y[M,Np] = epilogue( X[M,Kp](bf16) @ Wp[Np,Kp]^T + bias ), templated:
//   MODE 0: Y = relu6(acc + b)
//   MODE 1: Y = Res + relu6(acc + b)
//   MODE 2: Y = mask ? relu6(acc + b) : Res
// One wave32 per 16-row M-tile, MT tiles per wave; weights+bias in LDS.
// N-tiles processed in pairs -> two independent WMMA accumulator chains.
// Epilogue: C tile (col-per-lane) -> wave-private LDS -> row-per-lane b128 IO.

template <int KP, int NP, int MODE, int MT>
__global__ __launch_bounds__(256)
void gemm_kernel(const bf16* __restrict__ X, int ldx,
                 const bf16* __restrict__ Wp,
                 const float* __restrict__ bias,
                 bf16* __restrict__ Y, int ldy,
                 const bf16* __restrict__ Res, int ldr,
                 const unsigned char* __restrict__ mask) {
    extern __shared__ char smem[];
    bf16*  sW = (bf16*)smem;
    float* sB = (float*)(smem + (size_t)NP * KP * 2);
    float* sT = (float*)(smem + (size_t)NP * KP * 2 + (size_t)NP * 4);

    int tid = threadIdx.x;
    {   // cooperative stage of weights + bias into LDS (16B chunks)
        const uint4* s4 = (const uint4*)Wp;
        uint4*       d4 = (uint4*)sW;
        constexpr int N16 = NP * KP * 2 / 16;
        for (int i = tid; i < N16; i += 256) d4[i] = s4[i];
        for (int i = tid; i < NP;  i += 256) sB[i] = bias[i];
    }
    __syncthreads();

    int wave = tid >> 5, lane = tid & 31;
    int lr = lane & 15;   // A-row / B-col within tile
    int hi = lane >> 4;   // half-wave selector
    float* myT = sT + wave * (16 * 20);   // wave-private transpose pad

    constexpr int NKT = KP / 32;
    constexpr int NNT = NP / 16;
    static_assert((NNT & 1) == 0, "N-tile count must be even for pairing");

    for (int t = 0; t < MT; ++t) {
        int m0 = (((int)blockIdx.x * 8 + wave) * MT + t) * 16;
        unsigned mrow = (unsigned)(m0 + lr);
        const bf16* xrow = X + mrow * (unsigned)ldx;

        // Preload all A fragments (16-bit A layout: halves 0..7 -> K=hi*8+[0,8),
        // halves 8..15 -> K=16+hi*8+[0,8))
        v16bf A[NKT];
#pragma unroll
        for (int kt = 0; kt < NKT; ++kt) {
            v8bf a0 = *(const v8bf*)(xrow + kt * 32 + hi * 8);
            v8bf a1 = *(const v8bf*)(xrow + kt * 32 + 16 + hi * 8);
            v16bf a;
#pragma unroll
            for (int i = 0; i < 8; ++i) { a[i] = a0[i]; a[8 + i] = a1[i]; }
            A[kt] = a;
        }

        unsigned char mk = 0;
        if (MODE == 2) mk = mask[mrow];

        // Epilogue for one 16x16 tile: transpose via wave-private LDS pad.
        auto epilogue = [&](int nt, v8f acc) {
#pragma unroll
            for (int r = 0; r < 8; ++r) myT[(hi * 8 + r) * 20 + lr] = acc[r];
            __builtin_amdgcn_wave_barrier();   // DS ops in-order within wave
            const int cb = nt * 16 + hi * 8;
            float4 t0 = *(const float4*)(myT + lr * 20 + hi * 8);
            float4 t1 = *(const float4*)(myT + lr * 20 + hi * 8 + 4);
            float4 b0 = *(const float4*)(sB + cb);
            float4 b1 = *(const float4*)(sB + cb + 4);
            float tv[8] = { t0.x + b0.x, t0.y + b0.y, t0.z + b0.z, t0.w + b0.w,
                            t1.x + b1.x, t1.y + b1.y, t1.z + b1.z, t1.w + b1.w };
            v8bf resv = {};
            if (MODE != 0)
                resv = *(const v8bf*)(Res + mrow * (unsigned)ldr + (unsigned)cb);
            v8bf outv;
#pragma unroll
            for (int i = 0; i < 8; ++i) {
                float v = fminf(fmaxf(tv[i], 0.0f), 6.0f);   // ReLU6
                if (MODE == 1)      v += (float)resv[i];
                else if (MODE == 2) v = mk ? v : (float)resv[i];
                outv[i] = (bf16)v;
            }
            __builtin_amdgcn_wave_barrier();   // next tile's pad stores after reads
            *(v8bf*)(Y + mrow * (unsigned)ldy + (unsigned)cb) = outv;
        };

#pragma unroll
        for (int nt = 0; nt < NNT; nt += 2) {
            v8f acc0 = {}, acc1 = {};
#pragma unroll
            for (int kt = 0; kt < NKT; ++kt) {
                // B layout: n = lane%16, halves 0..7 -> K=hi*8+[0,8), halves 8..15 -> K=16+hi*8+[0,8)
                const bf16* wr0 = sW + ((nt + 0) * 16 + lr) * KP + kt * 32 + hi * 8;
                const bf16* wr1 = sW + ((nt + 1) * 16 + lr) * KP + kt * 32 + hi * 8;
                v8bf p0 = *(const v8bf*)wr0, p1 = *(const v8bf*)(wr0 + 16);
                v8bf q0 = *(const v8bf*)wr1, q1 = *(const v8bf*)(wr1 + 16);
                v16bf b0, b1;
#pragma unroll
                for (int i = 0; i < 8; ++i) {
                    b0[i] = p0[i]; b0[8 + i] = p1[i];
                    b1[i] = q0[i]; b1[8 + i] = q1[i];
                }
                // two independent chains -> WMMA hazard slots filled with work
                acc0 = __builtin_amdgcn_wmma_f32_16x16x32_bf16(
                    false, A[kt], false, b0, (short)0, acc0, false, false);
                acc1 = __builtin_amdgcn_wmma_f32_16x16x32_bf16(
                    false, A[kt], false, b1, (short)0, acc1, false, false);
            }
            epilogue(nt + 0, acc0);
            epilogue(nt + 1, acc1);
        }
    }
}

// ------------------- depthwise 3x3 + BN-affine (NHWC bf16) ------------------

template <int CS>
__global__ void dwconv_kernel(const bf16* __restrict__ X,
                              const bf16* __restrict__ dwp,   // [CS*9]
                              const float* __restrict__ g,    // [CS]
                              const float* __restrict__ bt,   // [CS]
                              bf16* __restrict__ Ytmp) {
    constexpr int CHUNKS = CS / 8;
    unsigned idx = blockIdx.x * blockDim.x + threadIdx.x;
    unsigned total = (unsigned)BHW_ * CHUNKS;
    if (idx >= total) return;
    unsigned pix = idx / CHUNKS;
    int  ch  = (int)(idx % CHUNKS) * 8;
    unsigned b = pix / HW_, hw = pix % HW_;
    int h = (int)(hw / W_), w = (int)(hw % W_);

    float acc[8];
#pragma unroll
    for (int i = 0; i < 8; ++i) acc[i] = 0.0f;

#pragma unroll
    for (int dy = -1; dy <= 1; ++dy) {
        int hh = h + dy;
        if ((unsigned)hh >= (unsigned)H_) continue;
#pragma unroll
        for (int dx = -1; dx <= 1; ++dx) {
            int ww = w + dx;
            if ((unsigned)ww >= (unsigned)W_) continue;
            v8bf xv = *(const v8bf*)(X + (b * HW_ + (unsigned)hh * W_ + (unsigned)ww) * CS + ch);
            int wi = (dy + 1) * 3 + (dx + 1);
#pragma unroll
            for (int i = 0; i < 8; ++i)
                acc[i] += (float)xv[i] * (float)dwp[(ch + i) * 9 + wi];
        }
    }
    v8bf out;
#pragma unroll
    for (int i = 0; i < 8; ++i)
        out[i] = (bf16)(acc[i] * g[ch + i] + bt[ch + i]);
    *(v8bf*)(Ytmp + pix * (unsigned)CS + ch) = out;
}

// --------------- final NHWC bf16 (stride 32, 17 valid) -> NCHW f32 ----------

__global__ void out_kernel(const bf16* __restrict__ Ybuf, float* __restrict__ out) {
    unsigned i = blockIdx.x * blockDim.x + threadIdx.x;
    unsigned total = (unsigned)B_ * 17u * HW_;
    if (i >= total) return;
    unsigned b = i / (17u * HW_);
    unsigned r = i % (17u * HW_);
    unsigned c = r / HW_;
    unsigned hw = r % HW_;
    out[i] = (float)Ybuf[(b * HW_ + hw) * 32u + c];
}

// ------------------------------- host side ----------------------------------

static inline size_t align_up(size_t x, size_t a) { return (x + a - 1) & ~(a - 1); }

extern "C" void kernel_launch(void* const* d_in, const int* in_sizes, int n_in,
                              void* d_out, int out_size, void* d_ws, size_t ws_size,
                              hipStream_t stream) {
    (void)in_sizes; (void)n_in; (void)out_size; (void)ws_size;

    const float* feat   = (const float*)d_in[0];
    const float* coarse = (const float*)d_in[1];

    // Input index map (JAX pytree: dict keys sorted; lists in order).
    // Per block base bb: b_in, b_mid0, b_mid1, b_out,
    //   conv0{bta,dw,g,pw}, conv1{...}, conv2{...}, w_in, w_mid0, w_mid1, w_out
    // block0 base = 2, block1 base = 22.

    char* ws = (char*)d_ws;
    size_t off = 0;
    auto alloc = [&](size_t bytes) -> char* {
        char* p = ws + off; off = align_up(off + bytes, 256); return p;
    };
    bf16* fmbuf = (bf16*)alloc((size_t)BHW_ * 96  * sizeof(bf16)); // fm (stride 96/64/32)
    bf16* ubuf  = (bf16*)alloc((size_t)BHW_ * 160 * sizeof(bf16)); // u / dw-conv tmp
    unsigned char* mask = (unsigned char*)alloc((size_t)BHW_);
    float* um = (float*)alloc((size_t)BHW_ * sizeof(float));

    auto packM = [&](int idx, int N, int K, int Np, int Kp) -> bf16* {
        bf16* p = (bf16*)alloc((size_t)Np * Kp * sizeof(bf16));
        unsigned tot = (unsigned)Np * Kp;
        pack_mat_kernel<<<(tot + 255) / 256, 256, 0, stream>>>(
            (const float*)d_in[idx], N, K, p, Np, Kp);
        return p;
    };
    auto packV = [&](const float* src, int N, int Np) -> float* {
        float* p = (float*)alloc((size_t)Np * sizeof(float));
        pack_vec_kernel<<<(Np + 255) / 256, 256, 0, stream>>>(src, N, p, Np);
        return p;
    };
    auto packDW = [&](int idx, int C, int Cp) -> bf16* {
        bf16* p = (bf16*)alloc((size_t)Cp * 9 * sizeof(bf16));
        int tot = Cp * 9;
        pack_dw_kernel<<<(tot + 255) / 256, 256, 0, stream>>>(
            (const float*)d_in[idx], C, p, Cp);
        return p;
    };
    auto F = [&](int idx) { return (const float*)d_in[idx]; };

    // ---- block0 packed params (C=70,U=140,cout=35 -> Cp=96,Up=160,Cp1=64) ----
    bf16* W_in0  = packM(18, 140,  70, 160,  96);
    bf16* W_m00  = packM(19, 140, 140, 160, 160);
    bf16* W_m01  = packM(20, 140, 140, 160, 160);
    bf16* W_out0 = packM(21,  70, 140,  96, 160);
    float* B_in0  = packV(F(2), 140, 160);
    float* B_m00  = packV(F(3), 140, 160);
    float* B_m01  = packV(F(4), 140, 160);
    float* B_out0 = packV(F(5),  70,  96);
    float* Zb     = packV(nullptr, 0, 160);          // zero bias (any Np<=160)
    bf16* DW00 = packDW(7, 70, 96);  float* G00 = packV(F(8), 70, 96);  float* T00 = packV(F(6), 70, 96);
    bf16* PW00 = packM(9,  70, 70, 96, 96);
    bf16* DW01 = packDW(11, 70, 96); float* G01 = packV(F(12), 70, 96); float* T01 = packV(F(10), 70, 96);
    bf16* PW01 = packM(13, 70, 70, 96, 96);
    bf16* DW02 = packDW(15, 70, 96); float* G02 = packV(F(16), 70, 96); float* T02 = packV(F(14), 70, 96);
    bf16* PW02 = packM(17, 35, 70, 64, 96);

    // ---- block1 packed params (C=35,U=70,cout=17 -> Cp=64,Up=96,CoutP=32) ----
    bf16* W_in1  = packM(38, 70, 35, 96, 64);
    bf16* W_m10  = packM(39, 70, 70, 96, 96);
    bf16* W_m11  = packM(40, 70, 70, 96, 96);
    bf16* W_out1 = packM(41, 35, 70, 64, 96);
    float* B_in1  = packV(F(22), 70, 96);
    float* B_m10  = packV(F(23), 70, 96);
    float* B_m11  = packV(F(24), 70, 96);
    float* B_out1 = packV(F(25), 35, 64);
    bf16* DW10 = packDW(27, 35, 64); float* G10 = packV(F(28), 35, 64); float* T10 = packV(F(26), 35, 64);
    bf16* PW10 = packM(29, 35, 35, 64, 64);
    bf16* DW11 = packDW(31, 35, 64); float* G11 = packV(F(32), 35, 64); float* T11 = packV(F(30), 35, 64);
    bf16* PW11 = packM(33, 35, 35, 64, 64);
    bf16* DW12 = packDW(35, 35, 64); float* G12 = packV(F(36), 35, 64); float* T12 = packV(F(34), 35, 64);
    bf16* PW12 = packM(37, 17, 35, 32, 64);

    // ---- uncertainty mask + initial feature pack ----
    unsigned nb1 = (unsigned)((BHW_ + 255) / 256);
    umax_kernel  <<<nb1, 256, 0, stream>>>(coarse, um);
    dilate_kernel<<<nb1, 256, 0, stream>>>(um, mask);
    {
        unsigned tot = (unsigned)BHW_ * 96u;
        pack_fm_kernel<<<(tot + 255) / 256, 256, 0, stream>>>(feat, coarse, fmbuf);
    }

    constexpr int MT = 4;                        // M-tiles per wave
    const unsigned GB = (unsigned)(BHW_ / (16 * 8 * MT));   // 1024 blocks
#define GEMM(KP, NP, MODE, Xp, LDX, WP, BP, YP, LDY, RP, LDR)                               \
    gemm_kernel<KP, NP, MODE, MT>                                                           \
        <<<GB, 256, (size_t)(NP) * (KP) * 2 + (NP) * 4 + 8 * 16 * 20 * 4, stream>>>(        \
            Xp, LDX, WP, BP, YP, LDY, RP, LDR, mask)

    auto dw96 = [&](const bf16* x, const bf16* dwp, const float* g, const float* bt, bf16* y) {
        unsigned tot = (unsigned)BHW_ * (96 / 8);
        dwconv_kernel<96><<<(tot + 255) / 256, 256, 0, stream>>>(x, dwp, g, bt, y);
    };
    auto dw64 = [&](const bf16* x, const bf16* dwp, const float* g, const float* bt, bf16* y) {
        unsigned tot = (unsigned)BHW_ * (64 / 8);
        dwconv_kernel<64><<<(tot + 255) / 256, 256, 0, stream>>>(x, dwp, g, bt, y);
    };

    // ================= block 0 =================
    GEMM(96, 160, 0, fmbuf, 96, W_in0,  B_in0,  ubuf, 160, (const bf16*)nullptr, 0);
    GEMM(160, 160, 1, ubuf, 160, W_m00, B_m00,  ubuf, 160, ubuf, 160);
    GEMM(160, 160, 1, ubuf, 160, W_m01, B_m01,  ubuf, 160, ubuf, 160);
    GEMM(160, 96, 2,  ubuf, 160, W_out0, B_out0, fmbuf, 96, fmbuf, 96);   // mask select
    dw96(fmbuf, DW00, G00, T00, ubuf);
    GEMM(96, 96, 1, ubuf, 96, PW00, Zb, fmbuf, 96, fmbuf, 96);            // residual
    dw96(fmbuf, DW01, G01, T01, ubuf);
    GEMM(96, 96, 1, ubuf, 96, PW01, Zb, fmbuf, 96, fmbuf, 96);            // residual
    dw96(fmbuf, DW02, G02, T02, ubuf);
    GEMM(96, 64, 0, ubuf, 96, PW02, Zb, fmbuf, 64, (const bf16*)nullptr, 0); // out conv

    // ================= block 1 =================
    GEMM(64, 96, 0, fmbuf, 64, W_in1,  B_in1,  ubuf, 96, (const bf16*)nullptr, 0);
    GEMM(96, 96, 1, ubuf, 96, W_m10, B_m10, ubuf, 96, ubuf, 96);
    GEMM(96, 96, 1, ubuf, 96, W_m11, B_m11, ubuf, 96, ubuf, 96);
    GEMM(96, 64, 2, ubuf, 96, W_out1, B_out1, fmbuf, 64, fmbuf, 64);      // mask select
    dw64(fmbuf, DW10, G10, T10, ubuf);
    GEMM(64, 64, 1, ubuf, 64, PW10, Zb, fmbuf, 64, fmbuf, 64);
    dw64(fmbuf, DW11, G11, T11, ubuf);
    GEMM(64, 64, 1, ubuf, 64, PW11, Zb, fmbuf, 64, fmbuf, 64);
    dw64(fmbuf, DW12, G12, T12, ubuf);
    GEMM(64, 32, 0, ubuf, 64, PW12, Zb, fmbuf, 32, (const bf16*)nullptr, 0);

    // ---- NHWC bf16 -> NCHW f32 output ----
    {
        unsigned tot = (unsigned)B_ * 17u * HW_;
        out_kernel<<<(tot + 255) / 256, 256, 0, stream>>>(fmbuf, (float*)d_out);
    }
#undef GEMM
}
